// ConvModel_23476291240061
// MI455X (gfx1250) — matrix operand order repaired and checked by reading, not compiled, not run on hardware
//
#include <hip/hip_runtime.h>

typedef __attribute__((ext_vector_type(16))) __bf16 v16bf;
typedef __attribute__((ext_vector_type(8)))  float  v8f;

#define C_IN   64
#define H_IMG  224
#define W_IMG  224
#define K_OUT  128
#define HW     (H_IMG * W_IMG)
#define KSTEPS 18                    // 9 taps * (64 channels / 32)
#define WELEMS (K_OUT * C_IN * 9)    // 73728 weight elements

// ---------------------------------------------------------------------------
// Prep: convert weights fp32 -> bf16 hi/lo once, pre-swizzled into per-lane
// WMMA A-fragment layout with K reordered as kNew = rs*64 + c.
//   ws layout: hi plane [wt][ks][lane][16 halves], then lo plane.
// ---------------------------------------------------------------------------
__global__ __launch_bounds__(256)
void prep_weights(const float* __restrict__ wgt,
                  __bf16* __restrict__ wsHi, __bf16* __restrict__ wsLo)
{
    const int idx = blockIdx.x * 256 + threadIdx.x;
    if (idx >= WELEMS) return;
    const int m   = idx / 576;
    const int rem = idx - m * 576;
    const int c   = rem / 9;
    const int rs  = rem - c * 9;
    const float v = wgt[idx];
    const __bf16 hi = (__bf16)v;
    const __bf16 lo = (__bf16)(v - (float)hi);
    // A (16x32, 16-bit) fragment layout (contiguous K, matches B staging):
    //  lanes 0-15 : halves 0-15 -> K 0-15
    //  lanes 16-31: halves 0-15 -> K 16-31
    const int ks    = rs * 2 + (c >> 5);
    const int k     = c & 31;
    const int wt    = m >> 4;
    const int laneA = (m & 15) + (k & 16);
    const int half  = k & 15;
    const int off   = (((wt * KSTEPS + ks) * 32 + laneA) << 4) + half;
    wsHi[off] = hi;
    wsLo[off] = lo;
}

// ---------------------------------------------------------------------------
// Main: implicit-GEMM conv, bf16x3 split precision (hi*hi + hi*lo + lo*hi,
// f32 accumulate). Block = 256 threads (8 wave32).
// Block tile: 128 (Mout) x 128 pixels (4 rows x 32 cols).
// Wave w: M rows [16w,16w+16) x all 8 pixel groups -> 8 f32 accumulators,
// 24 v_wmma_f32_16x16x32_bf16 per K-step.
// ---------------------------------------------------------------------------
__global__ __launch_bounds__(256)
void conv3x3_wmma_bf16x3(const float* __restrict__ x,
                         const __bf16* __restrict__ wsHi,
                         const __bf16* __restrict__ wsLo,
                         const float* __restrict__ bias,
                         float* __restrict__ out)
{
    __shared__ __align__(32) __bf16 sBhi[8][512];   // 8 pixel groups, 32 lanes x 16 halves
    __shared__ __align__(32) __bf16 sBlo[8][512];

    const int tid  = threadIdx.x;
    const int wave = tid >> 5;
    const int lane = tid & 31;

    // Grid: 16 images * 56 row-groups * 7 col-tiles
    const int b      = blockIdx.x;
    const int bw     = b % 7;
    const int t7     = b / 7;
    const int bh     = t7 % 56;
    const int nImg   = t7 / 56;
    const int owBase = bw * 32;
    const int ohBase = bh * 4;

    // Staging role: thread owns pixel n2 (fixed) and 16 consecutive K values.
    const int n2    = tid & 127;          // pixel within 4x32 tile
    const int row   = n2 >> 5;            // 0..3
    const int nw    = n2 & 31;            // 0..31
    const int kbase = (tid >> 7) << 4;    // 0 or 16
    const int pB    = row * 2 + (nw >> 4);            // pixel group 0..7
    const int laneB = (nw & 15) | kbase;              // (k&16) == kbase
    __bf16* const dHi = &sBhi[pB][laneB << 4];        // 16 halves contiguous (32B)
    __bf16* const dLo = &sBlo[pB][laneB << 4];

    v8f acc[8];
    #pragma unroll
    for (int p = 0; p < 8; ++p) acc[p] = (v8f){};

    for (int rs = 0; rs < 9; ++rs) {
        const int dr = rs / 3 - 1;          // unrolled? rs loop is short; SALU-cheap
        const int dc = rs - (rs / 3) * 3 - 1;
        const int ih = ohBase + row + dr;
        const int iw = owBase + nw + dc;
        const bool valid = ((unsigned)ih < H_IMG) & ((unsigned)iw < W_IMG);
        const float* const xbase =
            x + (((nImg * C_IN + kbase) * H_IMG + ih) * W_IMG + iw);

        #pragma unroll
        for (int cc = 0; cc < 2; ++cc) {
            // ---- stage B: 16 loads, one base + immediate offsets, one branch
            const float* const xp = xbase + cc * 32 * HW;
            float v[16];
            if (valid) {
                #pragma unroll
                for (int i = 0; i < 16; ++i) v[i] = xp[i * HW];
            } else {
                #pragma unroll
                for (int i = 0; i < 16; ++i) v[i] = 0.0f;
            }
            v16bf hv, lv;
            #pragma unroll
            for (int i = 0; i < 16; ++i) {
                const __bf16 h = (__bf16)v[i];
                hv[i] = h;
                lv[i] = (__bf16)(v[i] - (float)h);
            }
            *(v16bf*)dHi = hv;        // 2x ds_store_b128
            *(v16bf*)dLo = lv;
            __syncthreads();

            // ---- A fragments straight from L2 (preprocessed, pre-swizzled)
            const int ks   = rs * 2 + cc;
            const int aoff = ((wave * KSTEPS + ks) * 32 + lane) << 4;
            const v16bf aHi = *(const v16bf*)(wsHi + aoff);
            const v16bf aLo = *(const v16bf*)(wsLo + aoff);

            #pragma unroll
            for (int p = 0; p < 8; ++p) {
                const v16bf bh = *(const v16bf*)&sBhi[p][lane << 4];
                const v16bf bl = *(const v16bf*)&sBlo[p][lane << 4];
                acc[p] = __builtin_amdgcn_wmma_f32_16x16x32_bf16(
                             false, aHi, false, bh, (short)0, acc[p], false, false);
                acc[p] = __builtin_amdgcn_wmma_f32_16x16x32_bf16(
                             false, aHi, false, bl, (short)0, acc[p], false, false);
                acc[p] = __builtin_amdgcn_wmma_f32_16x16x32_bf16(
                             false, aLo, false, bh, (short)0, acc[p], false, false);
            }
            __syncthreads();          // LDS reused next K-step
        }
    }

    // C/D layout: VGPR r, lanes 0-15 -> M=r, lanes 16-31 -> M=r+8; N = lane&15.
    const int mBase = wave * 16 + ((lane >> 4) << 3);
    const int nl    = lane & 15;
    float bv[8];
    #pragma unroll
    for (int r = 0; r < 8; ++r) bv[r] = bias[mBase + r];

    #pragma unroll
    for (int p = 0; p < 8; ++p) {
        const int oh = ohBase + (p >> 1);
        const int ow = owBase + ((p & 1) << 4) + nl;
        #pragma unroll
        for (int r = 0; r < 8; ++r) {
            const int m = mBase + r;
            out[((nImg * K_OUT + m) * H_IMG + oh) * W_IMG + ow] = acc[p][r] + bv[r];
        }
    }
}

extern "C" void kernel_launch(void* const* d_in, const int* in_sizes, int n_in,
                              void* d_out, int out_size, void* d_ws, size_t ws_size,
                              hipStream_t stream) {
    (void)in_sizes; (void)n_in; (void)out_size; (void)ws_size;
    const float* x    = (const float*)d_in[0];
    const float* wgt  = (const float*)d_in[1];
    const float* bias = (const float*)d_in[2];
    float* out        = (float*)d_out;

    __bf16* wsHi = (__bf16*)d_ws;            // 73728 bf16
    __bf16* wsLo = wsHi + WELEMS;            // 73728 bf16  (total 288 KB scratch)

    prep_weights<<<dim3((WELEMS + 255) / 256), dim3(256), 0, stream>>>(wgt, wsHi, wsLo);

    const int nBlocks = 16 * (H_IMG / 4) * (W_IMG / 32);   // 6272
    conv3x3_wmma_bf16x3<<<dim3(nBlocks), dim3(256), 0, stream>>>(x, wsHi, wsLo, bias, out);
}
